// Matching_64845416235167
// MI455X (gfx1250) — matrix operand order repaired and checked by reading, not compiled, not run on hardware
//
#include <hip/hip_runtime.h>

typedef __bf16 bf16;
typedef bf16  v16bf __attribute__((ext_vector_type(16)));
typedef bf16  v8bf  __attribute__((ext_vector_type(8)));
typedef float v8f   __attribute__((ext_vector_type(8)));

constexpr int BN = 16, CH = 96, H = 128, W = 128, HW = H * W;

// ---------------------------------------------------------------------------
// 1) upflow: transposed conv (2,1,4,4), stride 2, groups=2 -> flow NCHW fp32
// ---------------------------------------------------------------------------
__global__ void k_upflow(const float* __restrict__ tflow,
                         const float* __restrict__ wu,
                         float* __restrict__ flow) {
  int idx = blockIdx.x * 256 + threadIdx.x;           // 524288 total
  int ox = idx & 127, oy = (idx >> 7) & 127, g = (idx >> 14) & 1, n = idx >> 15;
  float acc = 0.f;
#pragma unroll
  for (int ky = 0; ky < 4; ++ky) {
    int t = oy + ky - 2;
    if (t < 0 || (t & 1) || (t >> 1) >= 64) continue;
    int iy = t >> 1;
#pragma unroll
    for (int kx = 0; kx < 4; ++kx) {
      int s = ox + kx - 2;
      if (s < 0 || (s & 1) || (s >> 1) >= 64) continue;
      int ix = s >> 1;
      acc += tflow[((n * 2 + g) * 64 + iy) * 64 + ix] *
             wu[g * 16 + (3 - ky) * 4 + (3 - kx)];
    }
  }
  flow[idx] = acc;
}

// ---------------------------------------------------------------------------
// 2) backward warp with zero-border + validity mask. feat2 NCHW fp32.
// ---------------------------------------------------------------------------
__global__ void k_warp(const float* __restrict__ src,
                       const float* __restrict__ flow,
                       float* __restrict__ feat2) {
  int row = blockIdx.x;            // n*H + y
  int n = row >> 7, y = row & 127, x = threadIdx.x;
  float fx = (float)x + 2.5f * flow[((n * 2 + 0) * H + y) * W + x];
  float fy = (float)y + 2.5f * flow[((n * 2 + 1) * H + y) * W + x];
  float x0f = floorf(fx), y0f = floorf(fy);
  int ix0 = (int)x0f, iy0 = (int)y0f, ix1 = ix0 + 1, iy1 = iy0 + 1;
  float wa = (x0f + 1.f - fx) * (y0f + 1.f - fy);
  float wb = (fx - x0f) * (y0f + 1.f - fy);
  float wc = (x0f + 1.f - fx) * (fy - y0f);
  float wd = (fx - x0f) * (fy - y0f);
  bool vx0 = (ix0 >= 0) & (ix0 < W), vx1 = (ix1 >= 0) & (ix1 < W);
  bool vy0 = (iy0 >= 0) & (iy0 < H), vy1 = (iy1 >= 0) & (iy1 < H);
  float waE = (vx0 & vy0) ? wa : 0.f;
  float wbE = (vx1 & vy0) ? wb : 0.f;
  float wcE = (vx0 & vy1) ? wc : 0.f;
  float wdE = (vx1 & vy1) ? wd : 0.f;
  float wsum = waE + wbE + wcE + wdE;          // warped "ones" channel
  float mask = wsum > 0.999f ? 1.f : 0.f;
  waE *= mask; wbE *= mask; wcE *= mask; wdE *= mask;
  int cx0 = min(max(ix0, 0), W - 1), cx1 = min(max(ix1, 0), W - 1);
  int cy0 = min(max(iy0, 0), H - 1), cy1 = min(max(iy1, 0), H - 1);
  int oA = cy0 * W + cx0, oB = cy0 * W + cx1;
  int oC = cy1 * W + cx0, oD = cy1 * W + cx1;
  int base = n * CH * HW, oo = y * W + x;
  for (int c = 0; c < CH; ++c) {
    const float* p = src + base + c * HW;
    feat2[base + c * HW + oo] = waE * p[oA] + wbE * p[oB] + wcE * p[oC] + wdE * p[oD];
  }
}

// ---------------------------------------------------------------------------
// 3) 7x7 correlation (mean over 96 ch) + leaky-relu.
//    Output NHWC bf16 with 64 channels (49 real + 15 zero pad) so conv1's
//    K dim is a multiple of 32 -> clean WMMA chunks, 16B-aligned loads.
// ---------------------------------------------------------------------------
__global__ void k_corr(const float* __restrict__ first,
                       const float* __restrict__ feat2,
                       bf16* __restrict__ corr) {
  int row = blockIdx.x;
  int n = row >> 7, y = row & 127, x = threadIdx.x;
  float acc[49];
#pragma unroll
  for (int o = 0; o < 49; ++o) acc[o] = 0.f;
  int base = n * CH * HW;
  for (int c = 0; c < CH; ++c) {
    const float* p2 = feat2 + base + c * HW;
    float f = first[base + c * HW + y * W + x];
#pragma unroll
    for (int dy = 0; dy < 7; ++dy) {
      int iy = y + dy - 3;
      if (iy < 0 || iy >= H) continue;
      const float* pr = p2 + iy * W;
#pragma unroll
      for (int dx = 0; dx < 7; ++dx) {
        int ix = x + dx - 3;
        float v = (ix >= 0 && ix < W) ? pr[ix] : 0.f;
        acc[dy * 7 + dx] += f * v;
      }
    }
  }
  size_t ob = ((size_t)(n * H + y) * W + x) * 64;
#pragma unroll
  for (int o = 0; o < 49; ++o) {
    float r = acc[o] * (1.f / 96.f);
    r = r >= 0.f ? r : 0.1f * r;
    corr[ob + o] = (bf16)r;
  }
#pragma unroll
  for (int o = 49; o < 64; ++o) corr[ob + o] = (bf16)0.f;
}

// ---------------------------------------------------------------------------
// 4) weight pack: OIHW fp32 -> WMMA-B-fragment-swizzled bf16:
//    wdst[(chunk*OC + oc)*32 + kl], k = chunk*32+kl, k = tap*ICp + ic
//    (ic >= ICr zero-padded; lanes read two 8-bf16 runs, 16 elements apart)
// ---------------------------------------------------------------------------
__global__ void k_prepw(const float* __restrict__ wsrc, bf16* __restrict__ wdst,
                        int OC, int ICp, int ICr) {
  int i = blockIdx.x * 256 + threadIdx.x;
  int total = 9 * ICp * OC;
  if (i >= total) return;
  int kl = i & 31;
  int rest = i >> 5;
  int oc = rest % OC;
  int chunk = rest / OC;
  int k = chunk * 32 + kl;
  int tap = k / ICp, ic = k % ICp;
  int ky = tap / 3, kx = tap % 3;
  float v = (ic < ICr) ? wsrc[((oc * ICr + ic) * 3 + ky) * 3 + kx] : 0.f;
  wdst[i] = (bf16)v;
}

// ---------------------------------------------------------------------------
// 5) implicit-GEMM 3x3 conv, v_wmma_f32_16x16x32_bf16, IC % 32 == 0.
//    Each K-chunk of 32 sits inside one tap => A fragment = two contiguous
//    8-elem LDS runs (2x ds_load_b128), B fragment = two 8-elem runs (16 apart) from the
//    pre-swizzled weight buffer (2x global_load_b128). Fully unrolled.
// ---------------------------------------------------------------------------
template <int IC, int OC, int NW_N, bool RELU>
__global__ __launch_bounds__(256) void conv3x3_wmma(
    const bf16* __restrict__ in, const bf16* __restrict__ wpack,
    const float* __restrict__ bias, bf16* __restrict__ out) {
  static_assert(IC % 32 == 0, "IC must be a multiple of 32");
  constexpr int NW_M = 8 / NW_N;
  constexpr int TPX  = NW_M * 16;        // pixels per block (along x)
  constexpr int SCOL = TPX + 2;          // halo columns
  __shared__ __align__(16) bf16 smem[3 * SCOL * IC];

  const int row = blockIdx.y;            // n*H + y
  const int n = row >> 7, y = row & 127;
  const int x0 = blockIdx.x * TPX;
  const int ocb = blockIdx.z * (NW_N * 16);
  const int tid = threadIdx.x;

  if (tid == 0) __builtin_prefetch(wpack);   // global_prefetch_b8

  // Cooperative halo-tile load, vectorized by 8 bf16 (16B). Zero-pad borders.
  {
    constexpr int NV = IC / 8;                 // v8bf per (row,col)
    constexpr int TOT = 3 * SCOL * NV;
    v8bf* smemV = (v8bf*)smem;
    for (int i = tid; i < TOT; i += 256) {
      int icv = i % NV; int rest = i / NV;
      int col = rest % SCOL, r = rest / SCOL;
      int gy = y + r - 1, gx = x0 + col - 1;
      v8bf v = {};
      if (gy >= 0 && gy < H && gx >= 0 && gx < W)
        v = *(const v8bf*)(in + (((size_t)(n * H + gy) * W + gx) * IC + icv * 8));
      smemV[i] = v;
    }
  }
  __syncthreads();

  const int wid = tid >> 5, lane = tid & 31;
  const int wn = wid % NW_N, wm = wid / NW_N;
  const int half = lane >> 4, l16 = lane & 15;
  const int xA = wm * 16 + l16;               // A-matrix row (pixel) this lane owns
  const int oc = ocb + wn * 16 + l16;         // B/C column this lane owns

  // Per-lane base for B fragments (advances by OC*32 elements per chunk).
  const bf16* wp = wpack + (size_t)oc * 32 + half * 8;

  v8f acc = {};
  int chunk = 0;
#pragma unroll
  for (int tap = 0; tap < 9; ++tap) {
    const int ky = tap / 3, kx = tap % 3;
    // A base: lanes0-15 need K runs {0-7,16-23}, lanes16-31 {8-15,24-31}.
    const bf16* ap = smem + ((ky * SCOL) + (xA + kx)) * IC + half * 8;
#pragma unroll
    for (int ic0 = 0; ic0 < IC; ic0 += 32) {
      const v8bf* pa = (const v8bf*)(ap + ic0);
      v8bf alo = pa[0];                  // K run 1 (8 contiguous bf16)
      v8bf ahi = pa[2];                  // K run 2 (+16 elements)
      const v8bf* pb = (const v8bf*)(wp + (size_t)chunk * OC * 32);
      v8bf blo = pb[0];                  // K half*8 .. +7
      v8bf bhi = pb[2];                  // K 16+half*8 .. +7
      v16bf a = __builtin_shufflevector(alo, ahi, 0, 1, 2, 3, 4, 5, 6, 7,
                                        8, 9, 10, 11, 12, 13, 14, 15);
      v16bf b = __builtin_shufflevector(blo, bhi, 0, 1, 2, 3, 4, 5, 6, 7,
                                        8, 9, 10, 11, 12, 13, 14, 15);
      acc = __builtin_amdgcn_wmma_f32_16x16x32_bf16(false, a, false, b,
                                                    (short)0, acc, false, false);
      ++chunk;
    }
  }

  const float bv = bias[oc];
#pragma unroll
  for (int r = 0; r < 8; ++r) {
    int m = r + half * 8;                // C layout: VGPR r -> M=r (+8 hi lanes)
    int px = x0 + wm * 16 + m;
    float v = acc[r] + bv;
    if (RELU) v = v >= 0.f ? v : 0.1f * v;
    out[((size_t)(n * H + y) * W + px) * OC + oc] = (bf16)v;
  }
}

// ---------------------------------------------------------------------------
// 6) conv4 (32->2, 5x5, pad 2) fused with final flow + x (NCHW fp32 out)
// ---------------------------------------------------------------------------
__global__ void k_conv4(const bf16* __restrict__ x3, const float* __restrict__ w4,
                        const float* __restrict__ b4, const float* __restrict__ flow,
                        float* __restrict__ out) {
  int row = blockIdx.x;
  int n = row >> 7, y = row & 127, x = threadIdx.x;
  float a0 = b4[0], a1 = b4[1];
  for (int ky = 0; ky < 5; ++ky) {
    int iy = y + ky - 2;
    if (iy < 0 || iy >= H) continue;
    for (int kx = 0; kx < 5; ++kx) {
      int ix = x + kx - 2;
      if (ix < 0 || ix >= W) continue;
      const bf16* p = x3 + ((size_t)(n * H + iy) * W + ix) * 32;
      int wo = ky * 5 + kx;
#pragma unroll
      for (int ic = 0; ic < 32; ++ic) {
        float v = (float)p[ic];
        a0 += v * w4[ic * 25 + wo];
        a1 += v * w4[(32 + ic) * 25 + wo];
      }
    }
  }
  int o0 = ((n * 2 + 0) * H + y) * W + x;
  int o1 = ((n * 2 + 1) * H + y) * W + x;
  out[o0] = flow[o0] + a0;
  out[o1] = flow[o1] + a1;
}

// ---------------------------------------------------------------------------
extern "C" void kernel_launch(void* const* d_in, const int* in_sizes, int n_in,
                              void* d_out, int out_size, void* d_ws, size_t ws_size,
                              hipStream_t stream) {
  (void)in_sizes; (void)n_in; (void)out_size; (void)ws_size;
  const float* featFirst  = (const float*)d_in[2];
  const float* featSecond = (const float*)d_in[3];
  const float* tFlow      = (const float*)d_in[4];
  const float* wu         = (const float*)d_in[5];
  const float* w1 = (const float*)d_in[6];  const float* b1 = (const float*)d_in[7];
  const float* w2 = (const float*)d_in[8];  const float* b2 = (const float*)d_in[9];
  const float* w3 = (const float*)d_in[10]; const float* b3 = (const float*)d_in[11];
  const float* w4 = (const float*)d_in[12]; const float* b4 = (const float*)d_in[13];
  float* out = (float*)d_out;

  char* ws = (char*)d_ws;
  size_t cur = 0;
  auto alloc = [&](size_t bytes) {
    size_t o = cur;
    cur += (bytes + 255) & ~(size_t)255;
    return o;
  };
  float* flow  = (float*)(ws + alloc(sizeof(float) * BN * 2 * HW));     //   2 MB
  float* feat2 = (float*)(ws + alloc(sizeof(float) * BN * CH * HW));    // 100 MB
  bf16*  corr  = (bf16*) (ws + alloc(sizeof(bf16) * (size_t)BN * HW * 64));  // 33 MB
  bf16*  w1p   = (bf16*) (ws + alloc(sizeof(bf16) * 9 * 64 * 128));
  bf16*  w2p   = (bf16*) (ws + alloc(sizeof(bf16) * 9 * 128 * 64));
  bf16*  w3p   = (bf16*) (ws + alloc(sizeof(bf16) * 9 * 64 * 32));
  bf16*  x1    = (bf16*) (ws + alloc(sizeof(bf16) * (size_t)BN * HW * 128)); // 67 MB
  bf16*  x2    = (bf16*) (ws + alloc(sizeof(bf16) * (size_t)BN * HW * 64));  // 33 MB
  bf16*  x3    = (bf16*) (ws + alloc(sizeof(bf16) * (size_t)BN * HW * 32));  // 17 MB

  k_upflow<<<2048, 256, 0, stream>>>(tFlow, wu, flow);
  k_warp<<<2048, 128, 0, stream>>>(featSecond, flow, feat2);
  k_corr<<<2048, 128, 0, stream>>>(featFirst, feat2, corr);

  k_prepw<<<(9 * 64 * 128 + 255) / 256, 256, 0, stream>>>(w1, w1p, 128, 64, 49);
  k_prepw<<<(9 * 128 * 64 + 255) / 256, 256, 0, stream>>>(w2, w2p, 64, 128, 128);
  k_prepw<<<(9 * 64 * 32 + 255) / 256, 256, 0, stream>>>(w3, w3p, 32, 64, 64);

  // grid: (W/TPX, N*H, OC/(16*NW_N))
  conv3x3_wmma<64, 128, 4, true><<<dim3(4, BN * H, 2), 256, 0, stream>>>(corr, w1p, b1, x1);
  conv3x3_wmma<128, 64, 4, true><<<dim3(4, BN * H, 1), 256, 0, stream>>>(x1, w2p, b2, x2);
  conv3x3_wmma<64, 32, 2, true><<<dim3(2, BN * H, 1), 256, 0, stream>>>(x2, w3p, b3, x3);

  k_conv4<<<2048, 128, 0, stream>>>(x3, w4, b4, flow, out);
}
